// Self_Attention_17248588661488
// MI455X (gfx1250) — compile-verified, matrix operand order checked by reading, not run on hardware
//
#include <hip/hip_runtime.h>
#include <hip/hip_bf16.h>
#include <stdint.h>

// Problem constants (match reference)
#define Bn   4
#define Cn   256
#define Wn   4096
#define MALL 320   // 32 q + 32 k + 256 v output channels stacked

typedef __attribute__((ext_vector_type(16))) __bf16 v16bf;
typedef __attribute__((ext_vector_type(8)))  float  v8f;

union ABfrag { v16bf v; uint4 q[2]; };

__device__ __forceinline__ unsigned short f2bf(float f) {
  // round-to-nearest-even f32 -> bf16
  unsigned int u = __float_as_uint(f);
  u += 0x7FFFu + ((u >> 16) & 1u);
  return (unsigned short)(u >> 16);
}

__device__ __forceinline__ v8f wmma_bf16(const ABfrag& a, const ABfrag& b, v8f c) {
  // D = A(16x32 bf16) * B(32x16 bf16) + C(16x16 f32)
  return __builtin_amdgcn_wmma_f32_16x16x32_bf16(false, a.v, false, b.v,
                                                 (short)0, c, false, false);
}

// ---------------------------------------------------------------------------
// k0: pack Wq(32x256), Wk(32x256), Wv(256x256) -> Wall bf16 (320x256), ball f32
// ---------------------------------------------------------------------------
__global__ __launch_bounds__(256) void k0_pack(const float* Wq, const float* bq,
                                               const float* Wk, const float* bk,
                                               const float* Wv, const float* bv,
                                               unsigned short* wall, float* ball) {
  int idx = blockIdx.x * blockDim.x + threadIdx.x;   // 0..81919
  int m = idx >> 8, c = idx & 255;
  float w;
  if (m < 32)        w = Wq[m * Cn + c];
  else if (m < 64)   w = Wk[(m - 32) * Cn + c];
  else               w = Wv[(m - 64) * Cn + c];
  wall[m * Cn + c] = f2bf(w);
  if (idx < MALL) {
    float bb = (idx < 32) ? bq[idx] : (idx < 64 ? bk[idx - 32] : bv[idx - 64]);
    ball[idx] = bb;
  }
}

// ---------------------------------------------------------------------------
// k1: fused projection GEMM. P[m, w] = Wall[m,:] . x[b,:,w] + ball[m]
// M=320 (20 tiles), K=256, N=64 cols per workgroup. 10 waves, 2 M-tiles each.
// q -> qb (B,W,32) bf16 ; k -> kb (B,W,32) bf16 ; v -> vb (B,C,W) bf16
// ---------------------------------------------------------------------------
__global__ __launch_bounds__(320) void k1_proj(const float* __restrict__ x,
                                               const unsigned short* __restrict__ wall,
                                               const float* __restrict__ ball,
                                               unsigned short* qb, unsigned short* kb,
                                               unsigned short* vb) {
  __shared__ unsigned short xs[64 * 40];   // [w][c] bf16, stride 40 (16B aligned)
  const int b     = blockIdx.y;
  const int wbase = blockIdx.x * 64;
  const int tid   = threadIdx.x;
  const int wid = tid >> 5, lane = tid & 31, h = lane >> 4, ln = lane & 15;

  v8f acc[2][4] = {};

  for (int kc = 0; kc < Cn; kc += 32) {
    // stage x[b, kc..kc+31, wbase..wbase+63] -> LDS transposed bf16
    for (int idx = tid; idx < 2048; idx += 320) {
      int wl = idx & 63, c = idx >> 6;
      float xv = x[((size_t)b * Cn + (kc + c)) * Wn + wbase + wl];
      xs[wl * 40 + c] = f2bf(xv);
    }
    __syncthreads();

    ABfrag a[2], bf[4];
#pragma unroll
    for (int mi = 0; mi < 2; ++mi) {
      int row = wid * 32 + mi * 16 + ln;
      const unsigned short* ap = wall + row * Cn + kc;
      a[mi].q[0] = *(const uint4*)(ap + h * 8);
      a[mi].q[1] = *(const uint4*)(ap + 16 + h * 8);
    }
#pragma unroll
    for (int nt = 0; nt < 4; ++nt) {
      const unsigned short* bp = xs + (nt * 16 + ln) * 40;
      bf[nt].q[0] = *(const uint4*)(bp + h * 8);
      bf[nt].q[1] = *(const uint4*)(bp + 16 + h * 8);
    }
#pragma unroll
    for (int mi = 0; mi < 2; ++mi)
#pragma unroll
      for (int nt = 0; nt < 4; ++nt)
        acc[mi][nt] = wmma_bf16(a[mi], bf[nt], acc[mi][nt]);
    __syncthreads();
  }

  // epilogue: add bias, convert bf16, scatter to qb/kb/vb (band uniform per wave)
#pragma unroll
  for (int mi = 0; mi < 2; ++mi)
#pragma unroll
    for (int nt = 0; nt < 4; ++nt)
#pragma unroll
      for (int e = 0; e < 8; ++e) {
        int m = wid * 32 + mi * 16 + e + 8 * h;
        int w = wbase + nt * 16 + ln;
        unsigned short o = f2bf(acc[mi][nt][e] + ball[m]);
        if (m < 32)       qb[((size_t)b * Wn + w) * 32 + m] = o;
        else if (m < 64)  kb[((size_t)b * Wn + w) * 32 + (m - 32)] = o;
        else              vb[((size_t)b * Cn + (m - 64)) * Wn + w] = o;
      }
}

// ---------------------------------------------------------------------------
// k2: scores + exp. attn[b,row,col] = exp( q[b,row,:] . k[b,col,:] )
// One workgroup = (b, 16-row query tile); 8 waves cover 4096 cols (32 tiles ea).
// K=32 => exactly one v_wmma_f32_16x16x32_bf16 per 16x16 score tile.
// ---------------------------------------------------------------------------
__global__ __launch_bounds__(256) void k2_scores(const unsigned short* __restrict__ qb,
                                                 const unsigned short* __restrict__ kb,
                                                 float* __restrict__ attn) {
  const int b = blockIdx.y, qt = blockIdx.x;
  const int tid = threadIdx.x, wid = tid >> 5, lane = tid & 31, h = lane >> 4, ln = lane & 15;

  ABfrag a;
  {
    int row = qt * 16 + ln;
    const unsigned short* ap = qb + ((size_t)b * Wn + row) * 32;
    a.q[0] = *(const uint4*)(ap + h * 8);
    a.q[1] = *(const uint4*)(ap + 16 + h * 8);
  }
  float* ab = attn + (size_t)b * Wn * Wn;
#pragma unroll 4
  for (int t = 0; t < 32; ++t) {
    int col = wid * 512 + t * 16 + ln;
    const unsigned short* bp = kb + ((size_t)b * Wn + col) * 32;
    ABfrag bf;
    bf.q[0] = *(const uint4*)(bp + h * 8);
    bf.q[1] = *(const uint4*)(bp + 16 + h * 8);
    v8f c = {};
    c = wmma_bf16(a, bf, c);
    // scores are O(10): exp in f32 without max-subtraction is safe here
#pragma unroll
    for (int e = 0; e < 8; ++e) {
      int row = qt * 16 + e + 8 * h;
      ab[(size_t)row * Wn + col] = __expf(c[e]);
    }
  }
}

// ---------------------------------------------------------------------------
// k2b: per-row reciprocal sums of exp(scores). One workgroup per row, coalesced.
// ---------------------------------------------------------------------------
__global__ __launch_bounds__(256) void k2b_rowsum(const float* __restrict__ attn,
                                                  float* __restrict__ stats) {
  __shared__ float red[256];
  const size_t row = blockIdx.x;               // global row = b*W + j
  const float* p = attn + row * Wn;
  float s = 0.f;
  for (int i = threadIdx.x; i < Wn; i += 256) s += p[i];
  red[threadIdx.x] = s;
  __syncthreads();
  for (int off = 128; off > 0; off >>= 1) {
    if ((int)threadIdx.x < off) red[threadIdx.x] += red[threadIdx.x + off];
    __syncthreads();
  }
  if (threadIdx.x == 0) stats[row] = 1.0f / red[0];
}

// ---------------------------------------------------------------------------
// k3: normalize attention in place (final f32 output) + AV GEMM + epilogue.
// out[b,c,w] = gamma * sum_j v[b,c,j]*attn[b,j,w] + x[b,c,w]
// Workgroup = (b, 64-col block). 8 waves x (2 M-tiles x 4 N-tiles). K loop j.
// ---------------------------------------------------------------------------
__global__ __launch_bounds__(256) void k3_av(const unsigned short* __restrict__ vb,
                                             float* __restrict__ attn,
                                             const float* __restrict__ stats,
                                             const float* __restrict__ x,
                                             const float* __restrict__ gammap,
                                             float* __restrict__ out) {
  __shared__ unsigned short ps[64 * 40];   // [w][j] normalized bf16, stride 40
  const int b     = blockIdx.y;
  const int wbase = blockIdx.x * 64;
  const int tid = threadIdx.x, wid = tid >> 5, lane = tid & 31, h = lane >> 4, ln = lane & 15;
  const int mbase = wid * 32;

  float* attnb = attn + (size_t)b * Wn * Wn;
  const float* statb = stats + (size_t)b * Wn;

  v8f acc[2][4] = {};

  for (int jb = 0; jb < Wn; jb += 32) {
    // normalize this 32x64 attention block, write final f32 back, bf16 to LDS
    for (int idx = tid; idx < 2048; idx += 256) {
      int wl = idx & 63, j = idx >> 6;
      size_t ga = (size_t)(jb + j) * Wn + wbase + wl;
      float pv = attnb[ga] * statb[jb + j];
      attnb[ga] = pv;                         // final attention output
      ps[wl * 40 + j] = f2bf(pv);             // B operand (transposed to [w][j])
    }
    __syncthreads();

    ABfrag a[2], bf[4];
#pragma unroll
    for (int mi = 0; mi < 2; ++mi) {
      int c = mbase + mi * 16 + ln;
      const unsigned short* ap = vb + ((size_t)b * Cn + c) * Wn + jb;
      a[mi].q[0] = *(const uint4*)(ap + h * 8);
      a[mi].q[1] = *(const uint4*)(ap + 16 + h * 8);
    }
#pragma unroll
    for (int nt = 0; nt < 4; ++nt) {
      const unsigned short* bp = ps + (nt * 16 + ln) * 40;
      bf[nt].q[0] = *(const uint4*)(bp + h * 8);
      bf[nt].q[1] = *(const uint4*)(bp + 16 + h * 8);
    }
#pragma unroll
    for (int mi = 0; mi < 2; ++mi)
#pragma unroll
      for (int nt = 0; nt < 4; ++nt)
        acc[mi][nt] = wmma_bf16(a[mi], bf[nt], acc[mi][nt]);
    __syncthreads();
  }

  const float gamma = gammap[0];
#pragma unroll
  for (int mi = 0; mi < 2; ++mi)
#pragma unroll
    for (int nt = 0; nt < 4; ++nt)
#pragma unroll
      for (int e = 0; e < 8; ++e) {
        int c = mbase + mi * 16 + e + 8 * h;
        int w = wbase + nt * 16 + ln;
        size_t oi = ((size_t)b * Cn + c) * Wn + w;
        out[oi] = gamma * acc[mi][nt][e] + x[oi];
      }
}

// ---------------------------------------------------------------------------
// Workspace layout (bytes, all 16B aligned): total ~10.3 MB
// ---------------------------------------------------------------------------
#define OFF_WALL  0u            // 320*256*2   = 163840
#define OFF_BALL  163840u       // 320*4       = 1280
#define OFF_QB    165120u       // 4*4096*32*2 = 1048576
#define OFF_KB    1213696u      // 1048576
#define OFF_VB    2262272u      // 4*256*4096*2= 8388608
#define OFF_STATS 10650880u     // 4*4096*4    = 65536

extern "C" void kernel_launch(void* const* d_in, const int* in_sizes, int n_in,
                              void* d_out, int out_size, void* d_ws, size_t ws_size,
                              hipStream_t stream) {
  (void)in_sizes; (void)n_in; (void)out_size; (void)ws_size;
  const float* x     = (const float*)d_in[0];
  const float* Wq    = (const float*)d_in[1];
  const float* bq    = (const float*)d_in[2];
  const float* Wk    = (const float*)d_in[3];
  const float* bk    = (const float*)d_in[4];
  const float* Wv    = (const float*)d_in[5];
  const float* bv    = (const float*)d_in[6];
  const float* gamma = (const float*)d_in[7];

  float* out  = (float*)d_out;                       // (B, C, W)
  float* attn = out + (size_t)Bn * Cn * Wn;          // (B, W, W)

  char* ws = (char*)d_ws;
  unsigned short* wall  = (unsigned short*)(ws + OFF_WALL);
  float*          ball  = (float*)(ws + OFF_BALL);
  unsigned short* qb    = (unsigned short*)(ws + OFF_QB);
  unsigned short* kb    = (unsigned short*)(ws + OFF_KB);
  unsigned short* vb    = (unsigned short*)(ws + OFF_VB);
  float*          stats = (float*)(ws + OFF_STATS);

  k0_pack   <<<dim3(320),      dim3(256), 0, stream>>>(Wq, bq, Wk, bk, Wv, bv, wall, ball);
  k1_proj   <<<dim3(64, Bn),   dim3(320), 0, stream>>>(x, wall, ball, qb, kb, vb);
  k2_scores <<<dim3(Wn/16, Bn),dim3(256), 0, stream>>>(qb, kb, attn);
  k2b_rowsum<<<dim3(Bn*Wn),    dim3(256), 0, stream>>>(attn, stats);
  k3_av     <<<dim3(Wn/64, Bn),dim3(256), 0, stream>>>(vb, attn, stats, x, gamma, out);
}
